// RelPartialLearnableMultiHeadAttn_72172630442404
// MI455X (gfx1250) — matrix operand order repaired and checked by reading, not compiled, not run on hardware
//
#include <hip/hip_runtime.h>

// ---------------- problem constants ----------------
constexpr int NH   = 16;
constexpr int DH   = 64;
constexpr int DM   = 1024;
constexpr int QL   = 384;
constexpr int MEMN = 384;
constexpr int BS   = 16;
constexpr int KLN  = 768;          // KLEN
constexpr float SCALE = 0.125f;    // 1/sqrt(64)

// ---------------- vector types (trivially-copyable, union-safe) ----------------
typedef __attribute__((ext_vector_type(16))) __bf16 v16bf;
typedef __attribute__((ext_vector_type(8)))  float  v8f;
typedef __attribute__((ext_vector_type(4)))  unsigned int u32x4;
typedef __attribute__((ext_vector_type(4)))  float  f32x4;
typedef __attribute__((ext_vector_type(8)))  unsigned short u16x8;
typedef __attribute__((vector_size(16)))     int    i32x4v;   // async builtin param type

union FragU {
    u32x4 u[2];
    v16bf v;
    unsigned short h[16];
};

__device__ __forceinline__ unsigned short f2bf(float f) {
    return __builtin_bit_cast(unsigned short, (__bf16)f);   // HW cvt, RNE
}
__device__ __forceinline__ float bf2f(unsigned short h) {
    unsigned int u = ((unsigned int)h) << 16;
    return __builtin_bit_cast(float, u);
}
__device__ __forceinline__ v8f wmma_bf16(v16bf a, v16bf b, v8f c) {
    // D(16x16,f32) = A(16x32,bf16) x B(32x16,bf16) + C
    return __builtin_amdgcn_wmma_f32_16x16x32_bf16(false, a, false, b,
                                                   (short)0, c, false, false);
}

// ---------------- async global->LDS staging (guarded; falls back to plain copy) ----
#if __has_builtin(__builtin_amdgcn_global_load_async_to_lds_b128)
#define ASYNC_LDS 1
#endif

__device__ __forceinline__ void stage_b128(const unsigned short* g, unsigned short* l) {
#ifdef ASYNC_LDS
    __builtin_amdgcn_global_load_async_to_lds_b128((i32x4v*)g, (i32x4v*)l, 0, 0);
#else
    *(u32x4*)l = *(const u32x4*)g;
#endif
}
__device__ __forceinline__ void stage_wait() {
#ifdef ASYNC_LDS
#if __has_builtin(__builtin_amdgcn_s_wait_asynccnt)
    __builtin_amdgcn_s_wait_asynccnt(0);
#else
    asm volatile("s_wait_asynccnt 0x0" ::: "memory");
#endif
#endif
}

// =====================================================================
// Pre-conversion kernels (one-time per launch; removes cvt from GEMM hot loop
// and halves streamed bytes)
// =====================================================================

// straight f32 -> bf16, with optional two-source concat split (elements)
__global__ __launch_bounds__(256)
void cvt_bf16_kernel(const float* __restrict__ a, const float* __restrict__ b,
                     long long splitElems, unsigned short* __restrict__ out)
{
    size_t i8 = ((size_t)blockIdx.x * 256 + threadIdx.x) * 8;
    const float* src = ((long long)i8 < splitElems) ? (a + i8) : (b + (i8 - splitElems));
    f32x4 v0 = *(const f32x4*)(src);
    f32x4 v1 = *(const f32x4*)(src + 4);
    u16x8 o;
    o[0] = f2bf(v0.x); o[1] = f2bf(v0.y); o[2] = f2bf(v0.z); o[3] = f2bf(v0.w);
    o[4] = f2bf(v1.x); o[5] = f2bf(v1.y); o[6] = f2bf(v1.z); o[7] = f2bf(v1.w);
    *(u16x8*)(out + i8) = o;
}

// transpose + convert: in[K][N] f32 -> out[N][K] bf16 (32x32 LDS tiles)
__global__ __launch_bounds__(256)
void transpose_cvt_kernel(const float* __restrict__ in, unsigned short* __restrict__ out,
                          int K, int N)
{
    __shared__ unsigned short tile[32][33];
    int k0 = blockIdx.y * 32;
    int n0 = blockIdx.x * 32;
    int t  = threadIdx.x;
    int tn  = t & 31;
    int tk4 = t >> 5;          // 0..7
#pragma unroll
    for (int q = 0; q < 4; ++q) {
        int k = tk4 * 4 + q;
        tile[tn][k] = f2bf(in[(size_t)(k0 + k) * N + n0 + tn]);
    }
    __syncthreads();
    int tk  = t & 31;
    int tn4 = t >> 5;
#pragma unroll
    for (int q = 0; q < 4; ++q) {
        int n = tn4 * 4 + q;
        out[(size_t)(n0 + n) * K + k0 + tk] = tile[n][tk];
    }
}

// =====================================================================
// bf16 WMMA GEMM, A[M][1024] bf16, BT[N][1024] bf16 (pre-transposed),
// K=1024, 128x128 tiles, BK=64, async double-buffered LDS staging.
// mode 0: scatter q/k/vT epilogue (QKV projection)
// mode 1: RK[h][m][d] epilogue (r projection)
// mode 2: f32 + residual epilogue (output projection)
// =====================================================================
constexpr int BM = 128, BN = 128, BK = 64;
constexpr int LDS_STR = 72;               // halves; 144B rows, 16B aligned
constexpr int TILE_HALVES = BM * LDS_STR; // one A (or B) buffer

__global__ __launch_bounds__(256)
void gemm_bf16_kernel(const unsigned short* __restrict__ A,
                      const unsigned short* __restrict__ BT,
                      const float* __restrict__ Wres,
                      unsigned short* __restrict__ o_q, unsigned short* __restrict__ o_k,
                      unsigned short* __restrict__ o_vt, float* __restrict__ o_f32,
                      int M, int N, int mode)
{
    extern __shared__ char smem[];
    unsigned short* sAB = (unsigned short*)smem;  // [2][2][TILE_HALVES] : buf, A/B

    const int Kd   = 1024;
    const int t    = threadIdx.x;
    const int lane = t & 31;
    const int wv   = t >> 5;          // 0..7 waves
    const int wy   = wv >> 2;         // 0..1
    const int wx   = wv & 3;          // 0..3
    const int m0   = blockIdx.y * BM;
    const int n0   = blockIdx.x * BN;

    const int l15 = lane & 15;
    const int hiA = (lane < 16) ? 0 : 8;    // C row-half base
    const int hiB = (lane < 16) ? 0 : 16;   // B K base

    const v8f vzero = {0.f,0.f,0.f,0.f,0.f,0.f,0.f,0.f};
    v8f acc[4][2];
#pragma unroll
    for (int i = 0; i < 4; ++i)
#pragma unroll
        for (int j = 0; j < 2; ++j) acc[i][j] = vzero;

    // staging job for this thread: 4 chunks of 8 halves (16B) per matrix
    const int s_row = t >> 1;               // 0..127
    const int s_c   = (t & 1) * 32;         // halves 0 or 32

    auto issue_tile = [&](int k0, int buf) {
        unsigned short* dA = sAB + (size_t)buf * 2 * TILE_HALVES;
        unsigned short* dB = dA + TILE_HALVES;
#pragma unroll
        for (int q = 0; q < 4; ++q) {
            int c = s_c + q * 8;
            stage_b128(A  + (size_t)(m0 + s_row) * Kd + k0 + c, &dA[s_row * LDS_STR + c]);
            stage_b128(BT + (size_t)(n0 + s_row) * Kd + k0 + c, &dB[s_row * LDS_STR + c]);
        }
    };

    const int KT = Kd / BK;   // 16
    issue_tile(0, 0);
    for (int kt = 0; kt < KT; ++kt) {
        stage_wait();                 // this wave's async writes to LDS done
        __syncthreads();              // all waves' writes done; prev compute done
#ifdef ASYNC_LDS
        if (kt + 1 < KT) issue_tile((kt + 1) * BK, (kt + 1) & 1);
        const unsigned short* cA = sAB + (size_t)(kt & 1) * 2 * TILE_HALVES;
#else
        const unsigned short* cA = sAB;
#endif
        const unsigned short* cB = cA + TILE_HALVES;

#pragma unroll
        for (int ks = 0; ks < 2; ++ks) {
            const int kh = ks * 32;
            FragU bfr[2];
#pragma unroll
            for (int nf = 0; nf < 2; ++nf) {
                const unsigned short* p = &cB[(wx * 32 + nf * 16 + l15) * LDS_STR + kh + hiB];
                bfr[nf].u[0] = *(const u32x4*)(p);
                bfr[nf].u[1] = *(const u32x4*)(p + 8);
            }
            FragU afr[4];
#pragma unroll
            for (int mf = 0; mf < 4; ++mf) {
                const unsigned short* p = &cA[(wy * 64 + mf * 16 + l15) * LDS_STR + kh];
                afr[mf].u[0] = *(const u32x4*)(p + hiB);
                afr[mf].u[1] = *(const u32x4*)(p + hiB + 8);
            }
#pragma unroll
            for (int mf = 0; mf < 4; ++mf)
#pragma unroll
                for (int nf = 0; nf < 2; ++nf)
                    acc[mf][nf] = wmma_bf16(afr[mf].v, bfr[nf].v, acc[mf][nf]);
        }
#ifndef ASYNC_LDS
        __syncthreads();
        if (kt + 1 < KT) issue_tile((kt + 1) * BK, 0);
#endif
    }

    // ---- epilogue ----
#pragma unroll
    for (int mf = 0; mf < 4; ++mf) {
#pragma unroll
        for (int nf = 0; nf < 2; ++nf) {
#pragma unroll
            for (int vr = 0; vr < 8; ++vr) {
                int gm = m0 + wy * 64 + mf * 16 + vr + hiA;  // C rows: vr + (lane<16?0:8)
                int gn = n0 + wx * 32 + nf * 16 + l15;
                float val = acc[mf][nf][vr];
                if (mode == 0) {
                    int kk = gm / BS, b = gm % BS;
                    int sect = gn >> 10;
                    int h = (gn & 1023) >> 6;
                    int d = gn & 63;
                    if (sect == 0) {
                        if (kk >= MEMN)
                            o_q[(((size_t)(b * NH + h)) * QL + (kk - MEMN)) * DH + d] = f2bf(val);
                    } else if (sect == 1) {
                        o_k[(((size_t)(b * NH + h)) * KLN + kk) * DH + d] = f2bf(val);
                    } else {
                        o_vt[(((size_t)(b * NH + h)) * DH + d) * KLN + kk] = f2bf(val);
                    }
                } else if (mode == 1) {
                    int h = gn >> 6, d = gn & 63;
                    o_k[(((size_t)h) * KLN + gm) * DH + d] = f2bf(val);   // RK
                } else {
                    o_f32[(size_t)gm * N + gn] = val + Wres[(size_t)gm * N + gn];
                }
            }
        }
    }
}

// =====================================================================
// Bias split: QW = q + r_w_bias (in place), QR = q + r_r_bias
// =====================================================================
__global__ __launch_bounds__(256)
void bias_split_kernel(unsigned short* __restrict__ qw, unsigned short* __restrict__ qr,
                       const float* __restrict__ rwb, const float* __restrict__ rrb)
{
    int idx = blockIdx.x * 256 + threadIdx.x;   // over BS*NH*QL*DH
    int d = idx & 63;
    int h = (idx / (QL * DH)) & (NH - 1);
    float q = bf2f(qw[idx]);
    qw[idx] = f2bf(q + rwb[h * DH + d]);
    qr[idx] = f2bf(q + rrb[h * DH + d]);
}

// =====================================================================
// Fused rel-attention: per (q-block of 64, b, h); 4 waves x 16 q-rows.
// score[i,j] = (AC[i,j] + BD[i, j + QL-1-i]) * SCALE, masked j > i+MEM
// (rel_shift's wrap/zero region coincides exactly with the mask)
// =====================================================================
__global__ __launch_bounds__(128)
void attn_kernel(const unsigned short* __restrict__ QW,
                 const unsigned short* __restrict__ QR,
                 const unsigned short* __restrict__ KB,
                 const unsigned short* __restrict__ VT,
                 const unsigned short* __restrict__ RK,
                 unsigned short* __restrict__ AVB)
{
    extern __shared__ char smem[];
    float*          sS   = (float*)smem;                                   // [64][KLN] f32
    unsigned short* sBD  = (unsigned short*)(smem + (size_t)64 * KLN * 4); // [64][KLN] bf16
    float*          sSum = (float*)(smem + (size_t)64 * KLN * 6);          // [64]

    const int lane = threadIdx.x & 31;
    const int wv   = threadIdx.x >> 5;    // 0..3
    const int bq   = blockIdx.x;          // q block (of 64)
    const int b    = blockIdx.y;
    const int h    = blockIdx.z;
    const int l15  = lane & 15;
    const int hiC  = (lane < 16) ? 0 : 8;
    const int hiB  = (lane < 16) ? 0 : 16;
    const int rowBase = wv * 16;
    const int i0 = bq * 64 + rowBase;     // wave's global q-row base

    // ---- resident A-fragments for q+rwb and q+rrb (2 K-steps of 32) ----
    FragU aW[2], aR[2];
    {
        const size_t qoff = (((size_t)(b * NH + h)) * QL + i0 + l15) * DH;
        const unsigned short* qwp = QW + qoff;
        const unsigned short* qrp = QR + qoff;
#pragma unroll
        for (int ks = 0; ks < 2; ++ks) {
            int off = ks * 32 + ((lane < 16) ? 0 : 16);
            aW[ks].u[0] = *(const u32x4*)(qwp + off);
            aW[ks].u[1] = *(const u32x4*)(qwp + off + 8);
            aR[ks].u[0] = *(const u32x4*)(qrp + off);
            aR[ks].u[1] = *(const u32x4*)(qrp + off + 8);
        }
    }

    const v8f vzero = {0.f,0.f,0.f,0.f,0.f,0.f,0.f,0.f};

    // ---- Phase 1: BD = (q+rrb) . RK^T  -> sBD (bf16, unshifted) ----
    const unsigned short* rkp = RK + (size_t)h * KLN * DH;
    for (int mt = 0; mt < KLN / 16; ++mt) {
        v8f acc = vzero;
#pragma unroll
        for (int ks = 0; ks < 2; ++ks) {
            v16bf bv = *(const v16bf*)(rkp + (size_t)(mt * 16 + l15) * DH + ks * 32 + hiB);
            acc = wmma_bf16(aR[ks].v, bv, acc);
        }
#pragma unroll
        for (int vr = 0; vr < 8; ++vr)
            sBD[(rowBase + vr + hiC) * KLN + mt * 16 + l15] = f2bf(acc[vr]);
    }

    // ---- Phase 2: AC = (q+rwb) . K^T -> sS (f32) ----
    const unsigned short* kbp = KB + ((size_t)(b * NH + h)) * KLN * DH;
    for (int jt = 0; jt < KLN / 16; ++jt) {
        v8f acc = vzero;
#pragma unroll
        for (int ks = 0; ks < 2; ++ks) {
            v16bf bv = *(const v16bf*)(kbp + (size_t)(jt * 16 + l15) * DH + ks * 32 + hiB);
            acc = wmma_bf16(aW[ks].v, bv, acc);
        }
#pragma unroll
        for (int vr = 0; vr < 8; ++vr)
            sS[(rowBase + vr + hiC) * KLN + jt * 16 + l15] = acc[vr];
    }

    // ---- Phase 3: rel-shift merge + masked softmax (wave-local rows) ----
    {
        int r    = lane >> 1;
        int half = lane & 1;
        int row  = rowBase + r;
        int ig   = bq * 64 + row;
        int delta = QL - 1 - ig;        // shift for this row
        int jmax  = ig + MEMN;          // inclusive causal+mem bound
        float* srow = sS + (size_t)row * KLN;
        const unsigned short* brow = sBD + (size_t)row * KLN;

        float mx = -1e30f;
        for (int j = half; j < KLN; j += 2) {
            float v;
            if (j <= jmax) v = (srow[j] + bf2f(brow[j + delta])) * SCALE;
            else           v = -1e30f;
            srow[j] = v;
            mx = fmaxf(mx, v);
        }
        mx = fmaxf(mx, __shfl_xor(mx, 1, 32));

        float sum = 0.f;
        for (int j = half; j < KLN; j += 2) {
            float p = __expf(srow[j] - mx);
            srow[j] = p;                 // unnormalized prob
            sum += p;
        }
        sum += __shfl_xor(sum, 1, 32);
        if (half == 0) sSum[row] = sum;
    }

    // ---- Phase 4: O = P . V  (A from sS f32->bf16, B from VT, contiguous) ----
    v8f oacc[4];
#pragma unroll
    for (int nf = 0; nf < 4; ++nf) oacc[nf] = vzero;

    const unsigned short* vtp = VT + (((size_t)(b * NH + h)) * DH) * KLN;
    for (int js = 0; js < KLN / 32; ++js) {
        FragU ap;
        const float* prow = sS + (size_t)(rowBase + l15) * KLN + js * 32;
        int kb = (lane < 16) ? 0 : 16;
        f32x4 p0 = *(const f32x4*)(prow + kb);
        f32x4 p1 = *(const f32x4*)(prow + kb + 4);
        f32x4 p2 = *(const f32x4*)(prow + kb + 8);
        f32x4 p3 = *(const f32x4*)(prow + kb + 12);
        float tmp[16] = {p0.x,p0.y,p0.z,p0.w, p1.x,p1.y,p1.z,p1.w,
                         p2.x,p2.y,p2.z,p2.w, p3.x,p3.y,p3.z,p3.w};
#pragma unroll
        for (int e = 0; e < 16; ++e) ap.h[e] = f2bf(tmp[e]);

#pragma unroll
        for (int nf = 0; nf < 4; ++nf) {
            v16bf bv = *(const v16bf*)(vtp + (size_t)(nf * 16 + l15) * KLN + js * 32 + hiB);
            oacc[nf] = wmma_bf16(ap.v, bv, oacc[nf]);
        }
    }

    // ---- epilogue: normalize, write attn_vec bf16 [i*BS+b][h*64+d] ----
#pragma unroll
    for (int nf = 0; nf < 4; ++nf) {
#pragma unroll
        for (int vr = 0; vr < 8; ++vr) {
            int row = rowBase + vr + hiC;
            int ig  = bq * 64 + row;
            int d   = nf * 16 + l15;
            float val = oacc[nf][vr] / sSum[row];
            AVB[((size_t)ig * BS + b) * DM + h * DH + d] = f2bf(val);
        }
    }
}

// =====================================================================
// LayerNorm over last dim (1024), one WG (256 thr) per row
// =====================================================================
__global__ __launch_bounds__(256)
void layernorm_kernel(const float* __restrict__ X, const float* __restrict__ gamma,
                      const float* __restrict__ beta, float* __restrict__ out)
{
    __shared__ float red[16];
    int m = blockIdx.x;
    int t = threadIdx.x;
    const float* x = X + (size_t)m * DM;
    f32x4 v = *(const f32x4*)(x + t * 4);

    float s = v.x + v.y + v.z + v.w;
#pragma unroll
    for (int o = 16; o > 0; o >>= 1) s += __shfl_xor(s, o, 32);
    if ((t & 31) == 0) red[t >> 5] = s;
    __syncthreads();
    float tot = 0.f;
#pragma unroll
    for (int i = 0; i < 8; ++i) tot += red[i];
    float mu = tot * (1.0f / DM);

    float dx = v.x - mu, dy = v.y - mu, dz = v.z - mu, dw = v.w - mu;
    float vs = dx * dx + dy * dy + dz * dz + dw * dw;
#pragma unroll
    for (int o = 16; o > 0; o >>= 1) vs += __shfl_xor(vs, o, 32);
    if ((t & 31) == 0) red[8 + (t >> 5)] = vs;
    __syncthreads();
    float vtot = 0.f;
#pragma unroll
    for (int i = 0; i < 8; ++i) vtot += red[8 + i];
    float inv = rsqrtf(vtot * (1.0f / DM) + 1e-5f);

    f32x4 g  = *(const f32x4*)(gamma + t * 4);
    f32x4 bb = *(const f32x4*)(beta + t * 4);
    f32x4 o4;
    o4.x = dx * inv * g.x + bb.x;
    o4.y = dy * inv * g.y + bb.y;
    o4.z = dz * inv * g.z + bb.z;
    o4.w = dw * inv * g.w + bb.w;
    *(f32x4*)(out + (size_t)m * DM + t * 4) = o4;
}

// =====================================================================
extern "C" void kernel_launch(void* const* d_in, const int* in_sizes, int n_in,
                              void* d_out, int out_size, void* d_ws, size_t ws_size,
                              hipStream_t stream)
{
    const float* w     = (const float*)d_in[0];
    const float* r     = (const float*)d_in[1];
    const float* rwb   = (const float*)d_in[2];
    const float* rrb   = (const float*)d_in[3];
    const float* mems  = (const float*)d_in[4];
    // d_in[5] attn_mask: j > i + MEM, recomputed analytically in-kernel
    const float* Wqkv  = (const float*)d_in[6];
    const float* Wr    = (const float*)d_in[7];
    const float* Wo    = (const float*)d_in[8];
    const float* gamma = (const float*)d_in[9];
    const float* beta  = (const float*)d_in[10];

    char* ws = (char*)d_ws;
    size_t off = 0;
    auto alloc = [&](size_t bytes) -> void* {
        void* p = ws + off;
        off += (bytes + 255) & ~(size_t)255;
        return p;
    };
    // bf16 operand buffers
    unsigned short* CATB  = (unsigned short*)alloc((size_t)KLN * BS * DM * 2); // [12288][1024]
    unsigned short* RB    = (unsigned short*)alloc((size_t)KLN * DM * 2);      // [768][1024]
    unsigned short* WqkvT = (unsigned short*)alloc((size_t)3 * DM * DM * 2);   // [3072][1024]
    unsigned short* WrT   = (unsigned short*)alloc((size_t)DM * DM * 2);       // [1024][1024]
    unsigned short* WoT   = (unsigned short*)alloc((size_t)DM * DM * 2);       // [1024][1024]
    // attention operand buffers
    unsigned short* QW = (unsigned short*)alloc((size_t)BS * NH * QL * DH * 2);
    unsigned short* QR = (unsigned short*)alloc((size_t)BS * NH * QL * DH * 2);
    unsigned short* KB = (unsigned short*)alloc((size_t)BS * NH * KLN * DH * 2);
    unsigned short* VT = (unsigned short*)alloc((size_t)BS * NH * KLN * DH * 2);
    unsigned short* RK = (unsigned short*)alloc((size_t)NH * KLN * DH * 2);
    unsigned short* AVB = (unsigned short*)alloc((size_t)QL * BS * DM * 2);
    float* OP = (float*)alloc((size_t)QL * BS * DM * 4);

    const size_t gemm_lds = (size_t)2 * 2 * TILE_HALVES * 2;  // 73728 B ping-pong

    // 0) pre-convert operands to bf16 (and transpose weights to [N][K])
    cvt_bf16_kernel<<<(KLN * BS * DM) / (256 * 8), 256, 0, stream>>>(
        mems, w, (long long)MEMN * BS * DM, CATB);
    cvt_bf16_kernel<<<(KLN * DM) / (256 * 8), 256, 0, stream>>>(
        r, r, (long long)KLN * DM + 1, RB);
    {
        dim3 g1((3 * DM) / 32, DM / 32);
        transpose_cvt_kernel<<<g1, 256, 0, stream>>>(Wqkv, WqkvT, DM, 3 * DM);
        dim3 g2(DM / 32, DM / 32);
        transpose_cvt_kernel<<<g2, 256, 0, stream>>>(Wr, WrT, DM, DM);
        transpose_cvt_kernel<<<g2, 256, 0, stream>>>(Wo, WoT, DM, DM);
    }

    // 1) QKV projection: cat @ W_qkv  (M=12288, N=3072, K=1024)
    {
        dim3 grid((3 * DM) / BN, (KLN * BS) / BM);   // 24 x 96
        gemm_bf16_kernel<<<grid, 256, gemm_lds, stream>>>(CATB, WqkvT, nullptr,
            QW, KB, VT, nullptr, KLN * BS, 3 * DM, 0);
    }
    // 2) r @ W_r -> RK   (M=768, N=1024, K=1024)
    {
        dim3 grid(DM / BN, KLN / BM);                // 8 x 6
        gemm_bf16_kernel<<<grid, 256, gemm_lds, stream>>>(RB, WrT, nullptr,
            nullptr, RK, nullptr, nullptr, KLN, DM, 1);
    }
    // 3) bias split q -> q+rwb, q+rrb
    bias_split_kernel<<<(BS * NH * QL * DH) / 256, 256, 0, stream>>>(QW, QR, rwb, rrb);

    // 4) fused relative attention -> AVB bf16
    {
        dim3 grid(QL / 64, BS, NH);                  // 6 x 16 x 16
        size_t lds = (size_t)64 * KLN * 4 + (size_t)64 * KLN * 2 + 64 * 4; // ~288.25 KB
        attn_kernel<<<grid, 128, lds, stream>>>(QW, QR, KB, VT, RK, AVB);
    }
    // 5) output projection + residual: AV @ W_o + w  (M=6144, N=1024, K=1024)
    {
        dim3 grid(DM / BN, (QL * BS) / BM);          // 8 x 48
        gemm_bf16_kernel<<<grid, 256, gemm_lds, stream>>>(AVB, WoT, w,
            nullptr, nullptr, nullptr, OP, QL * BS, DM, 2);
    }
    // 6) layernorm -> d_out
    layernorm_kernel<<<QL * BS, 256, 0, stream>>>(OP, gamma, beta, (float*)d_out);
}
